// Regional_Proposal_Network_80831284511354
// MI455X (gfx1250) — matrix operand order repaired and checked by reading, not compiled, not run on hardware
//
#include <hip/hip_runtime.h>
#include <hip/hip_bf16.h>

typedef __attribute__((ext_vector_type(16))) __bf16 v16bf;
typedef __attribute__((ext_vector_type(8)))  float  v8f;

#define NIMG 8
#define CIN  512
#define COUT 256
#define FH   50
#define FW   50
#define NPIX 2500          // 50*50
#define KTOT 4608          // 512*9
#define NANCH 22500        // 2500*9
#define PRE_NMS_K 1000
#define MAX_PROP_K 1000

union FragU {
    uint4 q[2];
    v16bf v;
};

// ---------------------------------------------------------------------------
// f32 -> bf16 conversion of the feature map, transposed to [n][pos][ci]
// so im2col K-runs are contiguous loads.
__global__ void rpn_convert_feat(const float* __restrict__ f, __bf16* __restrict__ o) {
    int idx = blockIdx.x * blockDim.x + threadIdx.x;
    const int total = NIMG * CIN * NPIX;
    if (idx >= total) return;
    int ci  = idx % CIN;
    int pos = (idx / CIN) % NPIX;
    int n   = idx / (CIN * NPIX);
    o[idx] = (__bf16)f[((size_t)n * CIN + ci) * NPIX + pos];   // o: [n][pos][ci]
}

// conv_w (256,512,3,3) -> bf16 B^T matrix [co][k] with k = ci + 512*(ky*3+kx).
// Column-major weights keep both LDS staging and per-lane B fragments K-contiguous.
__global__ void rpn_convert_wt(const float* __restrict__ w, __bf16* __restrict__ o) {
    int idx = blockIdx.x * blockDim.x + threadIdx.x;
    if (idx >= KTOT * COUT) return;
    int k   = idx % KTOT;
    int co  = idx / KTOT;
    int ci  = k & 511;
    int tap = k >> 9;
    int ky = tap / 3, kx = tap % 3;
    o[idx] = (__bf16)w[(((size_t)co * CIN + ci) * 3 + ky) * 3 + kx];  // o: [co][k]
}

// ---------------------------------------------------------------------------
// 3x3 conv as im2col GEMM: per image, C[2500][256] = A[2500][4608] x B[4608][256]
// 128 threads = 4 wave32s. Tile: 128 (M) x 64 (N), K-step 64, K loop split as
// tap(9) x ci-chunk(8) so im2col addressing hoists out of the hot loop.
// Each wave: two 16-row strips x four 16-col tiles = 8 accumulators,
// 16 WMMAs per K-iteration; fragments preloaded so WMMAs issue back-to-back.
__global__ void __launch_bounds__(128)
rpn_conv3x3_wmma(const __bf16* __restrict__ Fbf,    // [NIMG][NPIX][CIN]
                 const __bf16* __restrict__ WbT,    // [COUT][KTOT]
                 const float*  __restrict__ convB,  // [COUT]
                 float* __restrict__ convOut)       // [NIMG][NPIX][COUT]
{
    __shared__ __align__(16) __bf16 As[128][72];  // [m-row][k], 144B stride (16B-mult)
    __shared__ __align__(16) __bf16 Bs[64][72];   // [n-col][k], 144B stride

    const int n      = blockIdx.z;
    const int mTile  = blockIdx.y * 128;
    const int coTile = blockIdx.x * 64;
    const int tid  = threadIdx.x;
    const int lane = tid & 31;
    const int wave = tid >> 5;
    const int g    = lane >> 4;     // lane half
    const int ln16 = lane & 15;

    v8f acc[2][4] = {};

    // A-tile load: one thread per row, 64 contiguous ci (128B = 8x uint4)
    const int ap = mTile + tid;                // pixel index for this thread's row
    const int ay = ap / FW, ax = ap % FW;
    // B-tile load: thread -> (col, 32-element half of the 64-wide K slab)
    const int br = tid >> 1;
    const int bh = tid & 1;
    const __bf16* wRow = WbT + (size_t)(coTile + br) * KTOT + bh * 32;

    #pragma unroll 1
    for (int tap = 0; tap < 9; ++tap) {
        const int ky = tap / 3, kx = tap % 3;
        const int iy = ay + ky - 1, ix = ax + kx - 1;
        const bool aValid = (ap < NPIX) & (iy >= 0) & (iy < FH) & (ix >= 0) & (ix < FW);
        const __bf16* aRow = Fbf + ((size_t)(n * NPIX + iy * FW + ix)) * CIN;
        const __bf16* wTap = wRow + tap * CIN;

        #pragma unroll 1
        for (int ciBase = 0; ciBase < CIN; ciBase += 64) {
            // ---- stage A (128 x 64) ----
            uint4* adst = (uint4*)&As[tid][0];
            if (aValid) {
                const uint4* src = (const uint4*)(aRow + ciBase);
                #pragma unroll
                for (int q = 0; q < 8; ++q) adst[q] = src[q];
            } else {
                const uint4 z = {0, 0, 0, 0};
                #pragma unroll
                for (int q = 0; q < 8; ++q) adst[q] = z;
            }
            // ---- stage B (64 x 64) ----
            {
                const uint4* src = (const uint4*)(wTap + ciBase);
                uint4* bdst = (uint4*)&Bs[br][bh * 32];
                #pragma unroll
                for (int q = 0; q < 4; ++q) bdst[q] = src[q];
            }
            if (ciBase + 64 < CIN)
                __builtin_prefetch(wTap + ciBase + 64, 0, 1);

            __syncthreads();

            // ---- two 32-wide K sub-steps ----
            #pragma unroll
            for (int kk = 0; kk < 64; kk += 32) {
                // Preload ALL fragments, then a single burst of 8 WMMAs.
                // A (16x32): lane m = lane%16; elem i -> K = kk + 8g+i / kk+16+8g+(i-8)
                FragU aF[2];
                #pragma unroll
                for (int s = 0; s < 2; ++s) {
                    const int mrow = wave * 32 + s * 16 + ln16;
                    aF[s].q[0] = *(const uint4*)&As[mrow][kk + 8 * g];
                    aF[s].q[1] = *(const uint4*)&As[mrow][kk + 16 + 8 * g];
                }
                // B (32x16): lane n = lane%16; elem i -> K = kk + 8g+i / kk+16+8g+(i-8)
                FragU bF[4];
                #pragma unroll
                for (int t = 0; t < 4; ++t) {
                    const int col = t * 16 + ln16;
                    bF[t].q[0] = *(const uint4*)&Bs[col][kk + 8 * g];
                    bF[t].q[1] = *(const uint4*)&Bs[col][kk + 16 + 8 * g];
                }
                #pragma unroll
                for (int s = 0; s < 2; ++s)
                    #pragma unroll
                    for (int t = 0; t < 4; ++t)
                        acc[s][t] = __builtin_amdgcn_wmma_f32_16x16x32_bf16(
                            false, aF[s].v, false, bF[t].v, (short)0, acc[s][t],
                            false, false);
            }
            __syncthreads();
        }
    }

    // C layout: VGPR v -> m = v + 8*g, n = lane%16
    #pragma unroll
    for (int s = 0; s < 2; ++s) {
        #pragma unroll
        for (int t = 0; t < 4; ++t) {
            const int co = coTile + t * 16 + ln16;
            #pragma unroll
            for (int v = 0; v < 8; ++v) {
                const int p = mTile + wave * 32 + s * 16 + v + 8 * g;
                if (p < NPIX)
                    convOut[((size_t)(n * NPIX + p)) * COUT + co] = acc[s][t][v] + convB[co];
            }
        }
    }
}

// ---------------------------------------------------------------------------
// 1x1 heads: 54 output channels (18 cls + 36 bbox), K = 256. Tiny GEMV per thread.
__global__ void rpn_heads(const float* __restrict__ convOut,
                          const float* __restrict__ clsW, const float* __restrict__ clsB,
                          const float* __restrict__ bbW,  const float* __restrict__ bbB,
                          float* __restrict__ clsOut,     // [N][18][NPIX]
                          float* __restrict__ bbOut)      // [N][36][NPIX]
{
    int idx = blockIdx.x * blockDim.x + threadIdx.x;
    const int total = NIMG * 54 * NPIX;
    if (idx >= total) return;
    int pos = idx % NPIX;
    int c   = (idx / NPIX) % 54;
    int n   = idx / (NPIX * 54);
    const float* f = convOut + ((size_t)(n * NPIX + pos)) * COUT;
    const float* w = (c < 18) ? (clsW + c * COUT) : (bbW + (c - 18) * COUT);
    float s = 0.f;
    #pragma unroll 4
    for (int ci = 0; ci < COUT; ++ci) s += f[ci] * w[ci];
    if (c < 18) clsOut[((size_t)n * 18 + c) * NPIX + pos] = s + clsB[c];
    else        bbOut [((size_t)n * 36 + (c - 18)) * NPIX + pos] = s + bbB[c - 18];
}

// ---------------------------------------------------------------------------
// Decode: anchors + bbox_adjust + clip + validity. Replicates the reference's
// NON-transposed cls indexing: score i <- channel (2i+1)/2500, pos (2i+1)%2500.
__global__ void rpn_decode(const float* __restrict__ clsOut, const float* __restrict__ bbOut,
                           float* __restrict__ boxes, float* __restrict__ sAll)
{
    int idx = blockIdx.x * blockDim.x + threadIdx.x;
    if (idx >= NIMG * NANCH) return;
    int i = idx % NANCH;
    int n = idx / NANCH;

    int f = 2 * i + 1;
    int c = f / NPIX, rem = f % NPIX;
    float logit = clsOut[((size_t)n * 18 + c) * NPIX + rem];
    float sc = 1.f / (1.f + __expf(-logit));

    int pos = i / 9, b = i % 9;
    int y = pos / FW, x = pos % FW;
    float d0 = bbOut[((size_t)n * 36 + b * 4 + 0) * NPIX + pos];
    float d1 = bbOut[((size_t)n * 36 + b * 4 + 1) * NPIX + pos];
    float d2 = bbOut[((size_t)n * 36 + b * 4 + 2) * NPIX + pos];
    float d3 = bbOut[((size_t)n * 36 + b * 4 + 3) * NPIX + pos];

    int r = b / 3, sI = b % 3;
    const float ratios[3] = {0.5f, 1.f, 2.f};
    const float sizes[3]  = {128.f, 256.f, 512.f};
    float hr = sqrtf(ratios[r]);
    float wr = 1.f / hr;
    float wsz = wr * sizes[sI], hsz = hr * sizes[sI];
    float bx1 = rintf(-wsz * 0.5f), by1 = rintf(-hsz * 0.5f);   // RNE = jnp.round
    float bx2 = rintf( wsz * 0.5f), by2 = rintf( hsz * 0.5f);
    float sx = x * 16.f, sy = y * 16.f;                         // 800//50 = 16
    float ax1 = sx + bx1, ay1 = sy + by1, ax2 = sx + bx2, ay2 = sy + by2;

    float aw = ax2 - ax1, ah = ay2 - ay1;
    float cx = ax1 + 0.5f * aw, cy = ay1 + 0.5f * ah;
    float ncx = d0 * aw + cx, ncy = d1 * ah + cy;
    float nw = __expf(d2) * aw, nh = __expf(d3) * ah;

    float x1 = fminf(fmaxf(ncx - 0.5f * nw, 0.f), 800.f);
    float y1 = fminf(fmaxf(ncy - 0.5f * nh, 0.f), 800.f);
    float x2 = fminf(fmaxf(ncx + 0.5f * nw, 0.f), 800.f);
    float y2 = fminf(fmaxf(ncy + 0.5f * nh, 0.f), 800.f);

    bool valid = (x2 - x1 >= 16.f) && (y2 - y1 >= 16.f) && (sc > 0.05f);
    float* bo = boxes + (size_t)idx * 4;
    bo[0] = x1; bo[1] = y1; bo[2] = x2; bo[3] = y2;
    sAll[idx] = valid ? sc : -1.f;
}

// ---------------------------------------------------------------------------
// Exact top-1000 via rank counting (stable ties on index, like lax.top_k).
// One workgroup per image; candidate scores streamed through a 16KB LDS tile.
__global__ void __launch_bounds__(1024)
rpn_topk(const float* __restrict__ sAll, const float* __restrict__ boxesAll,
         float* __restrict__ topv, float* __restrict__ topb)
{
    const int n = blockIdx.x;
    const int tid = threadIdx.x;
    const float* s = sAll + (size_t)n * NANCH;
    __shared__ float tile[4096];
    const int PER = (NANCH + 1023) / 1024;   // 22

    float myS[22];
    int   myCnt[22];
    #pragma unroll
    for (int e = 0; e < PER; ++e) {
        int i = tid + e * 1024;
        myS[e]   = (i < NANCH) ? s[i] : 0.f;
        myCnt[e] = (i < NANCH) ? 0 : (1 << 30);
    }
    for (int base = 0; base < NANCH; base += 4096) {
        for (int k = tid; k < 4096; k += 1024) {
            int j = base + k;
            tile[k] = (j < NANCH) ? s[j] : -3.0e38f;  // below any real score; never counted
        }
        __syncthreads();
        #pragma unroll
        for (int e = 0; e < PER; ++e) {
            int i = tid + e * 1024;
            if (i < NANCH) {
                float sv = myS[e];
                int cacc = 0;
                for (int jj = 0; jj < 4096; ++jj) {
                    float o = tile[jj];
                    cacc += ((o > sv) || (o == sv && (base + jj) < i)) ? 1 : 0;
                }
                myCnt[e] += cacc;
            }
        }
        __syncthreads();
    }
    #pragma unroll
    for (int e = 0; e < PER; ++e) {
        int i = tid + e * 1024;
        if (i < NANCH && myCnt[e] < PRE_NMS_K) {
            int r = myCnt[e];
            topv[(size_t)n * PRE_NMS_K + r] = myS[e];
            const float* b = boxesAll + ((size_t)n * NANCH + i) * 4;
            float* d = topb + ((size_t)n * PRE_NMS_K + r) * 4;
            d[0] = b[0]; d[1] = b[1]; d[2] = b[2]; d[3] = b[3];
        }
    }
}

// ---------------------------------------------------------------------------
// Greedy NMS, boxes in LDS, sequential over i (matches the fori_loop semantics).
__global__ void __launch_bounds__(1024)
rpn_nms(const float* __restrict__ topb, int* __restrict__ keep)
{
    const int n = blockIdx.x;
    const int tid = threadIdx.x;
    __shared__ float4 bs[PRE_NMS_K];
    __shared__ int    ks[PRE_NMS_K];
    for (int i = tid; i < PRE_NMS_K; i += 1024) {
        bs[i] = ((const float4*)topb)[n * PRE_NMS_K + i];
        ks[i] = 1;
    }
    __syncthreads();
    for (int i = 0; i < PRE_NMS_K - 1; ++i) {
        if (ks[i]) {                                   // uniform: same LDS word for all lanes
            float4 bi = bs[i];
            float areai = (bi.z - bi.x) * (bi.w - bi.y);
            for (int j = tid; j < PRE_NMS_K; j += 1024) {
                if (j > i && ks[j]) {
                    float4 bj = bs[j];
                    float lx = fmaxf(bi.x, bj.x), ly = fmaxf(bi.y, bj.y);
                    float rx = fminf(bi.z, bj.z), ry = fminf(bi.w, bj.w);
                    float iw = fmaxf(rx - lx, 0.f), ih = fmaxf(ry - ly, 0.f);
                    float inter = iw * ih;
                    float areaj = (bj.z - bj.x) * (bj.w - bj.y);
                    float iou = inter / (areai + areaj - inter + 1e-8f);
                    if (iou > 0.7f) ks[j] = 0;
                }
            }
        }
        __syncthreads();
    }
    for (int i = tid; i < PRE_NMS_K; i += 1024) keep[n * PRE_NMS_K + i] = ks[i];
}

// ---------------------------------------------------------------------------
// Final: s2 = keep && topv>0 ? topv : -1; full stable sort of 1000 via rank
// counting; write all 1000 output rows (box or zeros) + batch index column.
__global__ void __launch_bounds__(1024)
rpn_final(const float* __restrict__ topv, const float* __restrict__ topb,
          const int* __restrict__ keep, float* __restrict__ out)
{
    const int n = blockIdx.x;
    const int tid = threadIdx.x;
    __shared__ float s2[PRE_NMS_K];
    for (int i = tid; i < PRE_NMS_K; i += 1024) {
        float v = topv[n * PRE_NMS_K + i];
        s2[i] = (keep[n * PRE_NMS_K + i] && v > 0.f) ? v : -1.f;
    }
    __syncthreads();
    if (tid < PRE_NMS_K) {
        const int i = tid;
        float sv = s2[i];
        int r = 0;
        for (int j = 0; j < PRE_NMS_K; ++j) {
            float o = s2[j];
            r += ((o > sv) || (o == sv && j < i)) ? 1 : 0;
        }
        float* dst = out + ((size_t)n * MAX_PROP_K + r) * 5;
        const float* b = topb + ((size_t)n * PRE_NMS_K + i) * 4;
        bool pos = sv > 0.f;
        dst[0] = pos ? b[0] : 0.f;
        dst[1] = pos ? b[1] : 0.f;
        dst[2] = pos ? b[2] : 0.f;
        dst[3] = pos ? b[3] : 0.f;
        dst[4] = (float)n;
    }
}

// ---------------------------------------------------------------------------
extern "C" void kernel_launch(void* const* d_in, const int* in_sizes, int n_in,
                              void* d_out, int out_size, void* d_ws, size_t ws_size,
                              hipStream_t stream) {
    (void)in_sizes; (void)n_in; (void)out_size; (void)ws_size;
    // inputs: image_list, feature_map, conv_w, conv_b, cls_w, cls_b, bbox_w, bbox_b
    const float* feat  = (const float*)d_in[1];
    const float* convW = (const float*)d_in[2];
    const float* convB = (const float*)d_in[3];
    const float* clsW  = (const float*)d_in[4];
    const float* clsB  = (const float*)d_in[5];
    const float* bbW   = (const float*)d_in[6];
    const float* bbB   = (const float*)d_in[7];

    char* base = (char*)d_ws;
    size_t off = 0;
    auto carve = [&](size_t bytes) {
        char* p = base + off;
        off = (off + bytes + 255) & ~(size_t)255;
        return p;
    };
    __bf16* Fbf    = (__bf16*)carve((size_t)NIMG * NPIX * CIN * 2);
    __bf16* WbT    = (__bf16*)carve((size_t)COUT * KTOT * 2);
    float* convOut = (float*)carve((size_t)NIMG * NPIX * COUT * 4);
    float* clsOut  = (float*)carve((size_t)NIMG * 18 * NPIX * 4);
    float* bbOut   = (float*)carve((size_t)NIMG * 36 * NPIX * 4);
    float* boxes   = (float*)carve((size_t)NIMG * NANCH * 4 * 4);
    float* sAll    = (float*)carve((size_t)NIMG * NANCH * 4);
    float* topv    = (float*)carve((size_t)NIMG * PRE_NMS_K * 4);
    float* topb    = (float*)carve((size_t)NIMG * PRE_NMS_K * 4 * 4);
    int*   keep    = (int*)carve((size_t)NIMG * PRE_NMS_K * 4);

    {
        int tot = NIMG * CIN * NPIX;
        rpn_convert_feat<<<(tot + 255) / 256, 256, 0, stream>>>(feat, Fbf);
    }
    {
        int tot = KTOT * COUT;
        rpn_convert_wt<<<(tot + 255) / 256, 256, 0, stream>>>(convW, WbT);
    }
    {
        dim3 g(COUT / 64, (NPIX + 127) / 128, NIMG);   // (4, 20, 8)
        rpn_conv3x3_wmma<<<g, 128, 0, stream>>>(Fbf, WbT, convB, convOut);
    }
    {
        int tot = NIMG * 54 * NPIX;
        rpn_heads<<<(tot + 255) / 256, 256, 0, stream>>>(convOut, clsW, clsB, bbW, bbB,
                                                         clsOut, bbOut);
    }
    {
        int tot = NIMG * NANCH;
        rpn_decode<<<(tot + 255) / 256, 256, 0, stream>>>(clsOut, bbOut, boxes, sAll);
    }
    rpn_topk<<<NIMG, 1024, 0, stream>>>(sAll, boxes, topv, topb);
    rpn_nms<<<NIMG, 1024, 0, stream>>>(topb, keep);
    rpn_final<<<NIMG, 1024, 0, stream>>>(topv, topb, keep, (float*)d_out);
}
